// Seq2Seq_79199196938730
// MI455X (gfx1250) — matrix operand order repaired and checked by reading, not compiled, not run on hardware
//
#include <hip/hip_runtime.h>
#include <hip/hip_bf16.h>

// ---------------------------------------------------------------------------
// Types for CDNA5 WMMA bf16 path
// ---------------------------------------------------------------------------
typedef __attribute__((ext_vector_type(16))) __bf16 v16bf;
typedef __attribute__((ext_vector_type(8)))  float  v8f;

union Frag {
  v16bf v;
  uint4 q[2];   // 32 bytes
};

__device__ inline unsigned short f2bf(float x) {
  unsigned u = __float_as_uint(x);
  unsigned r = u + 0x7FFFu + ((u >> 16) & 1u);   // round-to-nearest-even
  return (unsigned short)(r >> 16);
}

__device__ inline float sigmoidf(float x) { return 1.0f / (1.0f + __expf(-x)); }

// ---------------------------------------------------------------------------
// Weight prep: pack [Wih | Whh] rows into bf16 [1024, 384]
// ---------------------------------------------------------------------------
__global__ void prep_lstm_w(const float* __restrict__ Wih,   // [1024,128]
                            const float* __restrict__ Whh,   // [1024,256]
                            unsigned short* __restrict__ dst) // [1024,384] bf16
{
  int i = blockIdx.x * blockDim.x + threadIdx.x;
  if (i >= 1024 * 384) return;
  int r = i / 384, k = i % 384;
  float v = (k < 128) ? Wih[r * 128 + k] : Whh[r * 256 + (k - 128)];
  dst[i] = f2bf(v);
}

__global__ void cvt_bf16(const float* __restrict__ src,
                         unsigned short* __restrict__ dst, int n)
{
  int i = blockIdx.x * blockDim.x + threadIdx.x;
  if (i < n) dst[i] = f2bf(src[i]);
}

// ---------------------------------------------------------------------------
// Init: h0=c0=0, tok = trg[:,0], argmax slots, threefry-2x32 teacher mask
// (jax.random.key(1) -> key (0,1); uniform = bits>>9|0x3f800000 - 1 < 0.7)
// ---------------------------------------------------------------------------
__device__ inline unsigned rotl32(unsigned x, int r) { return (x << r) | (x >> (32 - r)); }

__device__ inline void threefry2x32(unsigned k0, unsigned k1, unsigned& x0, unsigned& x1) {
  const int R0[4] = {13, 15, 26, 6};
  const int R1[4] = {17, 29, 16, 24};
  unsigned ks[3] = {k0, k1, 0x1BD11BDAu ^ k0 ^ k1};
  x0 += ks[0]; x1 += ks[1];
  for (int g = 0; g < 5; ++g) {
    const int* R = (g & 1) ? R1 : R0;
    for (int r = 0; r < 4; ++r) { x0 += x1; x1 = rotl32(x1, R[r]); x1 ^= x0; }
    x0 += ks[(g + 1) % 3];
    x1 += ks[(g + 2) % 3] + (unsigned)(g + 1);
  }
}

__global__ void init_state(const int* __restrict__ trg,
                           unsigned short* __restrict__ h0,
                           float* __restrict__ c,
                           int* __restrict__ tok,
                           int* __restrict__ teacher,
                           unsigned long long* __restrict__ slots)
{
  int tid = threadIdx.x;  // 256 threads
  for (int i = tid; i < 64 * 256; i += 256) { h0[i] = 0; c[i] = 0.0f; }
  for (int i = tid; i < 64 * 63; i += 256) slots[i] = 0ull;
  if (tid < 64) tok[tid] = trg[tid * 64 + 0];
  if (tid < 32) {
    // counts = [0..62] padded with one 0 -> 64 entries split in halves
    unsigned x0 = (unsigned)tid;
    unsigned x1 = (32 + tid < 63) ? (unsigned)(32 + tid) : 0u;
    threefry2x32(0u, 1u, x0, x1);
    float f0 = __uint_as_float((x0 >> 9) | 0x3f800000u) - 1.0f;
    teacher[tid] = (f0 < 0.7f) ? 1 : 0;
    if (32 + tid < 63) {
      float f1 = __uint_as_float((x1 >> 9) | 0x3f800000u) - 1.0f;
      teacher[32 + tid] = (f1 < 0.7f) ? 1 : 0;
    }
  }
}

// ---------------------------------------------------------------------------
// Fused LSTM cell step.  grid = (16 h_tiles, 4 b_tiles), block = 128 (4 waves)
// Wave w computes gate w's 16x16 tile of gates = [x|h] @ [Wih|Whh]^T via
// 12 x v_wmma_f32_16x16x32_bf16 (K = 384), then the block does the pointwise
// c/h update through LDS.  h is ping-ponged across launches (bf16).
// ---------------------------------------------------------------------------
#define XSTR 392   // padded LDS row stride (bf16 elems) for 384-wide X

__global__ __launch_bounds__(128) void lstm_step(
    const int* __restrict__ tok_ptr, int tok_stride, int tok_off,
    const float* __restrict__ emb,          // [V,128] f32
    const unsigned short* __restrict__ W,   // [1024,384] bf16  (rows: i,f,g,o)
    const float* __restrict__ bias,         // [1024] f32
    const unsigned short* __restrict__ h_in,  // [64,256] bf16
    unsigned short* __restrict__ h_out,       // [64,256] bf16
    float* __restrict__ c)                    // [64,256] f32 (in place, block-exclusive)
{
  __shared__ unsigned short Xs[16 * XSTR];        // 16 batch rows x K=384
  __shared__ float G[4][16][16];                  // gate pre-activations

  const int htile = blockIdx.x;        // 0..15
  const int b0    = blockIdx.y * 16;   // batch tile
  const int tid   = threadIdx.x;

  // Stage X = [embed(token) | h_prev] as bf16 into LDS
  for (int i = tid; i < 16 * 128; i += 128) {
    int r = i >> 7, k = i & 127;
    int token = tok_ptr[(b0 + r) * tok_stride + tok_off];
    Xs[r * XSTR + k] = f2bf(emb[(size_t)token * 128 + k]);
  }
  for (int i = tid; i < 16 * 256; i += 128) {
    int r = i >> 8, k = i & 255;
    Xs[r * XSTR + 128 + k] = h_in[(b0 + r) * 256 + k];
  }
  __syncthreads();

  const int w    = tid >> 5;      // gate index (i,f,g,o)
  const int lane = tid & 31;
  const int mrow = lane & 15;
  const int hi   = (lane < 16) ? 0 : 1;

  // A fragment: row mrow, K chunks {kb..kb+7} U {kb+16..kb+23}, kb = hi?8:0
  const unsigned short* Arow = Xs + mrow * XSTR + (hi ? 8 : 0);
  // B fragment: W row (= output col), K chunks {kb..kb+7} U {kb+16..kb+23}, kb = hi?8:0
  const unsigned short* Brow =
      W + (size_t)(w * 256 + htile * 16 + mrow) * 384 + (hi ? 8 : 0);

  v8f acc = {};
#pragma unroll
  for (int kk = 0; kk < 12; ++kk) {
    Frag a, b;
    a.q[0] = *(const uint4*)(Arow + kk * 32);
    a.q[1] = *(const uint4*)(Arow + kk * 32 + 16);
    b.q[0] = *(const uint4*)(Brow + kk * 32);
    b.q[1] = *(const uint4*)(Brow + kk * 32 + 16);
    acc = __builtin_amdgcn_wmma_f32_16x16x32_bf16(
        false, a.v, false, b.v, (short)0, acc, false, false);
  }

  // C layout: VGPR j -> row (j + 8*hi), col = lane&15
#pragma unroll
  for (int j = 0; j < 8; ++j) G[w][j + 8 * hi][mrow] = acc[j];
  __syncthreads();

  // Pointwise LSTM update for this (b_tile, h_tile) block
  for (int e = tid; e < 256; e += 128) {
    int r = e >> 4, hc = e & 15;
    int gidx = htile * 16 + hc;
    float gi = G[0][r][hc] + bias[gidx];
    float gf = G[1][r][hc] + bias[256 + gidx];
    float gg = G[2][r][hc] + bias[512 + gidx];
    float go = G[3][r][hc] + bias[768 + gidx];
    size_t idx = (size_t)(b0 + r) * 256 + gidx;
    float cn = sigmoidf(gf) * c[idx] + sigmoidf(gi) * tanhf(gg);
    c[idx] = cn;
    h_out[idx] = f2bf(sigmoidf(go) * tanhf(cn));
  }
}

// ---------------------------------------------------------------------------
// FC projection + fused argmax.  grid = (250, 4), block = 256 (8 waves).
// Wave w computes a 16x16 logits tile (K=256 -> 8 WMMA), adds bias, stores to
// d_out, and folds a per-row argmax via shuffles + packed u64 atomicMax.
// ---------------------------------------------------------------------------
__global__ __launch_bounds__(256) void fc_step(
    const unsigned short* __restrict__ h,     // [64,256] bf16
    const unsigned short* __restrict__ fcW,   // [32000,256] bf16
    const float* __restrict__ fcb,            // [32000] f32
    float* __restrict__ out,                  // [64,63,32000] f32
    unsigned long long* __restrict__ slots,   // [63,64] packed (key<<32)|~idx
    int t)
{
  __shared__ unsigned short Hs[16 * 264];     // staged h tile, padded rows

  const int b0  = blockIdx.y * 16;
  const int tid = threadIdx.x;

  // Stage 16x256 bf16 h tile (512 x uint4)
  for (int i = tid; i < 512; i += 256) {
    int r = i >> 5, kc = (i & 31) * 8;
    *(uint4*)(Hs + r * 264 + kc) = *(const uint4*)(h + (b0 + r) * 256 + kc);
  }
  __syncthreads();

  const int w     = tid >> 5;
  const int lane  = tid & 31;
  const int mrow  = lane & 15;
  const int hi    = (lane < 16) ? 0 : 1;
  const int ntile = blockIdx.x * 8 + w;
  const int vcol  = ntile * 16 + mrow;        // vocab column for this lane

  const unsigned short* Arow = Hs + mrow * 264 + (hi ? 8 : 0);
  const unsigned short* Brow = fcW + (size_t)vcol * 256 + (hi ? 8 : 0);

  v8f acc = {};
#pragma unroll
  for (int kk = 0; kk < 8; ++kk) {
    Frag a, b;
    a.q[0] = *(const uint4*)(Arow + kk * 32);
    a.q[1] = *(const uint4*)(Arow + kk * 32 + 16);
    b.q[0] = *(const uint4*)(Brow + kk * 32);
    b.q[1] = *(const uint4*)(Brow + kk * 32 + 16);
    acc = __builtin_amdgcn_wmma_f32_16x16x32_bf16(
        false, a.v, false, b.v, (short)0, acc, false, false);
  }

  const float bval = fcb[vcol];
#pragma unroll
  for (int j = 0; j < 8; ++j) {
    float v = acc[j] + bval;
    int brow = b0 + 8 * hi + j;
    out[((size_t)brow * 63 + t) * 32000 + vcol] = v;

    // per-row argmax over the 16 lanes of this half-wave
    float mv = v; int mc = vcol;
    for (int m = 8; m >= 1; m >>= 1) {
      float ov = __shfl_xor(mv, m, 16);
      int   oc = __shfl_xor(mc, m, 16);
      if (ov > mv || (ov == mv && oc < mc)) { mv = ov; mc = oc; }
    }
    if (mrow == 0) {
      unsigned u = __float_as_uint(mv);
      unsigned key = (u & 0x80000000u) ? ~u : (u | 0x80000000u);
      unsigned long long pk =
          ((unsigned long long)key << 32) | (unsigned)(0xFFFFFFFFu - (unsigned)mc);
      atomicMax(&slots[t * 64 + brow], pk);
    }
  }
}

// ---------------------------------------------------------------------------
// Token selection: teacher forcing vs decoded argmax
// ---------------------------------------------------------------------------
__global__ void next_tok_k(const unsigned long long* __restrict__ slots,
                           const int* __restrict__ trg,
                           const int* __restrict__ teacher,
                           int* __restrict__ tok, int t)
{
  int b = threadIdx.x;
  if (b < 64) {
    unsigned low = (unsigned)(slots[t * 64 + b] & 0xFFFFFFFFull);
    int am = (int)(0xFFFFFFFFu - low);
    tok[b] = teacher[t] ? trg[b * 64 + (t + 1)] : am;
  }
}

// ---------------------------------------------------------------------------
// Host launch
// ---------------------------------------------------------------------------
extern "C" void kernel_launch(void* const* d_in, const int* in_sizes, int n_in,
                              void* d_out, int out_size, void* d_ws, size_t ws_size,
                              hipStream_t stream)
{
  (void)in_sizes; (void)n_in; (void)out_size; (void)ws_size;

  const int*   src     = (const int*)  d_in[0];
  const int*   trg     = (const int*)  d_in[1];
  const float* enc_emb = (const float*)d_in[2];
  const float* enc_Wih = (const float*)d_in[3];
  const float* enc_Whh = (const float*)d_in[4];
  const float* enc_b   = (const float*)d_in[5];
  const float* dec_emb = (const float*)d_in[6];
  const float* dec_Wih = (const float*)d_in[7];
  const float* dec_Whh = (const float*)d_in[8];
  const float* dec_b   = (const float*)d_in[9];
  const float* fc_W    = (const float*)d_in[10];
  const float* fc_b    = (const float*)d_in[11];
  float* out = (float*)d_out;

  char* ws = (char*)d_ws;
  size_t off = 0;
  auto take = [&](size_t bytes) -> char* {
    char* p = ws + off;
    off = (off + bytes + 255) & ~(size_t)255;
    return p;
  };
  unsigned short* encW = (unsigned short*)take((size_t)1024 * 384 * 2);
  unsigned short* decW = (unsigned short*)take((size_t)1024 * 384 * 2);
  unsigned short* fcWb = (unsigned short*)take((size_t)32000 * 256 * 2);
  unsigned short* h0   = (unsigned short*)take((size_t)64 * 256 * 2);
  unsigned short* h1   = (unsigned short*)take((size_t)64 * 256 * 2);
  float*          cbuf = (float*)take((size_t)64 * 256 * 4);
  int*            tok  = (int*)take(64 * 4);
  int*            tch  = (int*)take(64 * 4);
  unsigned long long* slots = (unsigned long long*)take((size_t)64 * 63 * 8);

  // --- weight prep (bf16) ---
  prep_lstm_w<<<(1024 * 384 + 255) / 256, 256, 0, stream>>>(enc_Wih, enc_Whh, encW);
  prep_lstm_w<<<(1024 * 384 + 255) / 256, 256, 0, stream>>>(dec_Wih, dec_Whh, decW);
  cvt_bf16<<<32000, 256, 0, stream>>>(fc_W, fcWb, 32000 * 256);
  init_state<<<1, 256, 0, stream>>>(trg, h0, cbuf, tok, tch, slots);

  unsigned short* hb[2] = {h0, h1};
  int cur = 0;

  // --- encoder: 128 sequential LSTM steps ---
  for (int t = 0; t < 128; ++t) {
    lstm_step<<<dim3(16, 4), 128, 0, stream>>>(
        src, /*stride=*/128, /*off=*/t, enc_emb, encW, enc_b,
        hb[cur], hb[cur ^ 1], cbuf);
    cur ^= 1;
  }

  // --- decoder: 63 autoregressive steps ---
  for (int t = 0; t < 63; ++t) {
    lstm_step<<<dim3(16, 4), 128, 0, stream>>>(
        tok, /*stride=*/1, /*off=*/0, dec_emb, decW, dec_b,
        hb[cur], hb[cur ^ 1], cbuf);
    cur ^= 1;
    fc_step<<<dim3(250, 4), 256, 0, stream>>>(hb[cur], fcWb, fc_b, out, slots, t);
    if (t < 62)
      next_tok_k<<<1, 64, 0, stream>>>(slots, trg, tch, tok, t);
  }
}
